// InternImageLayer_dcnv3_18897856103314
// MI455X (gfx1250) — matrix operand order repaired and checked by reading, not compiled, not run on hardware
//
#include <hip/hip_runtime.h>
#include <hip/hip_bf16.h>
#include <math.h>

// ---------------------------------------------------------------------------
// InternImage DCNv3 block for MI455X (gfx1250, wave32, WMMA 16x16x32 f16)
// ---------------------------------------------------------------------------

typedef _Float16 half_t;
typedef __attribute__((ext_vector_type(16))) _Float16 v16h;
typedef __attribute__((ext_vector_type(8)))  _Float16 v8h;
typedef __attribute__((ext_vector_type(8)))  float    v8f;

#define NB      4
#define HDIM    96
#define WDIM    96
#define C_DIM   128
#define GRP     8
#define GC_DIM  16
#define PPTS    9
#define NOFF    144          // G*P*2
#define NMSK    72           // G*P
#define NOM     216          // NOFF + NMSK
#define HID_DIM 512
#define NTOK    (NB*HDIM*WDIM)   // 36864

__device__ __forceinline__ float gelu_exact(float x) {
    return 0.5f * x * (1.0f + erff(x * 0.70710678118654752440f));
}

// ---- weight convert+transpose: out[n*Kd + k] = (f16) in[k*Nd + n] ----------
__global__ void k_transpose_f16(const float* __restrict__ in, half_t* __restrict__ out,
                                int Kd, int Nd) {
    int i = blockIdx.x * blockDim.x + threadIdx.x;
    if (i >= Kd * Nd) return;
    int n = i / Kd, k = i - n * Kd;
    out[i] = (half_t)in[(size_t)k * Nd + n];
}

__global__ void k_concat_bias(const float* __restrict__ b0, const float* __restrict__ b1,
                              float* __restrict__ out) {
    int i = threadIdx.x;
    if (i < NOFF)      out[i] = b0[i];
    else if (i < NOM)  out[i] = b1[i - NOFF];
}

// ---- LayerNorm over C=128, one wave per token (lane holds 4 channels) ------
__global__ void k_layernorm(const float* __restrict__ in, const float* __restrict__ gam,
                            const float* __restrict__ bet, float* __restrict__ outF,
                            half_t* __restrict__ outH) {
    int wave = threadIdx.x >> 5, lane = threadIdx.x & 31;
    int tok = blockIdx.x * 8 + wave;
    const float4 v = *(const float4*)(in + (size_t)tok * C_DIM + lane * 4);
    float s = v.x + v.y + v.z + v.w;
    float q = v.x*v.x + v.y*v.y + v.z*v.z + v.w*v.w;
#pragma unroll
    for (int o = 16; o > 0; o >>= 1) { s += __shfl_xor(s, o, 32); q += __shfl_xor(q, o, 32); }
    float mean = s * (1.0f / C_DIM);
    float var  = q * (1.0f / C_DIM) - mean * mean;
    float rstd = rsqrtf(var + 1e-5f);
    float vv[4] = {v.x, v.y, v.z, v.w};
#pragma unroll
    for (int i = 0; i < 4; ++i) {
        int c = lane * 4 + i;
        float y = (vv[i] - mean) * rstd * gam[c] + bet[c];
        size_t idx = (size_t)tok * C_DIM + c;
        if (outF) outF[idx] = y;
        if (outH) outH[idx] = (half_t)y;
    }
}

// ---- depthwise 3x3 SAME + bias + LN + exact GELU, one wave per token -------
__global__ void k_dwconv_ln_gelu(const float* __restrict__ xin, const float* __restrict__ kw,
                                 const float* __restrict__ kb, const float* __restrict__ gam,
                                 const float* __restrict__ bet, half_t* __restrict__ outH) {
    int wave = threadIdx.x >> 5, lane = threadIdx.x & 31;
    int tok = blockIdx.x * 8 + wave;
    int n   = tok / (HDIM * WDIM);
    int rem = tok - n * HDIM * WDIM;
    int h = rem / WDIM, w = rem - h * WDIM;
    float acc[4];
#pragma unroll
    for (int i = 0; i < 4; ++i) acc[i] = kb[lane * 4 + i];
#pragma unroll
    for (int ky = 0; ky < 3; ++ky) {
        int yy = h + ky - 1;
        if (yy < 0 || yy >= HDIM) continue;
#pragma unroll
        for (int kx = 0; kx < 3; ++kx) {
            int xx = w + kx - 1;
            if (xx < 0 || xx >= WDIM) continue;
            const float4 sv = *(const float4*)(xin + (((size_t)n * HDIM + yy) * WDIM + xx) * C_DIM + lane * 4);
            const float4 kv = *(const float4*)(kw + (ky * 3 + kx) * C_DIM + lane * 4);
            acc[0] += sv.x * kv.x; acc[1] += sv.y * kv.y;
            acc[2] += sv.z * kv.z; acc[3] += sv.w * kv.w;
        }
    }
    float s = acc[0] + acc[1] + acc[2] + acc[3];
    float q = acc[0]*acc[0] + acc[1]*acc[1] + acc[2]*acc[2] + acc[3]*acc[3];
#pragma unroll
    for (int o = 16; o > 0; o >>= 1) { s += __shfl_xor(s, o, 32); q += __shfl_xor(q, o, 32); }
    float mean = s * (1.0f / C_DIM);
    float var  = q * (1.0f / C_DIM) - mean * mean;
    float rstd = rsqrtf(var + 1e-5f);
#pragma unroll
    for (int i = 0; i < 4; ++i) {
        int c = lane * 4 + i;
        float y = (acc[i] - mean) * rstd * gam[c] + bet[c];
        outH[(size_t)tok * C_DIM + c] = (half_t)gelu_exact(y);
    }
}

// ---- WMMA GEMM: out = act(A[MxK,f16] * Bt[NxK,f16]^T + bias) (+resid) ------
// Wave computes 16x64 tile; block = 8 waves -> 128x64; grid=(ceil(N/64), M/128)
// Out-of-range columns are CLAMPED (not branched) in the K-loop so the inner
// loop is branch-free; the store guard discards the clamped columns.
__global__ void k_gemm_wmma(const half_t* __restrict__ A, const half_t* __restrict__ Bt,
                            const float* __restrict__ bias, const float* __restrict__ resid,
                            float* __restrict__ outF, half_t* __restrict__ outH,
                            int Md, int Nd, int Kd, int act) {
    const int tid  = threadIdx.x;
    const int wave = tid >> 5, lane = tid & 31;
    const int r16  = lane & 15, hi = lane >> 4;
    const int m0   = blockIdx.y * 128 + wave * 16;
    const int n0   = blockIdx.x * 64;

    v8f acc[4];
#pragma unroll
    for (int j = 0; j < 4; ++j)
#pragma unroll
        for (int i = 0; i < 8; ++i) acc[j][i] = 0.0f;

    // CDNA5 16-bit A 16x32 layout: lanes0-15 K(0..7,16..23); lanes16-31 K(8..15,24..31)
    const half_t* arow = A + (size_t)(m0 + r16) * Kd + hi * 8;

    // CDNA5 16-bit B 32x16 layout (matches A): lanes0-15 K(0..7,16..23); lanes16-31 K(8..15,24..31).
    // Clamp invalid columns to a valid row of Bt (results discarded on store).
    const half_t* bptr[4];
#pragma unroll
    for (int j = 0; j < 4; ++j) {
        int col  = n0 + j * 16 + r16;
        int colc = (col < Nd) ? col : (Nd - 1);
        bptr[j] = Bt + (size_t)colc * Kd + hi * 8;
    }

    for (int k0 = 0; k0 < Kd; k0 += 32) {
        __builtin_prefetch(arow + k0 + 32, 0, 1);   // speculative; drops past end
        v8h alo = *(const v8h*)(arow + k0);
        v8h ahi = *(const v8h*)(arow + k0 + 16);
        v16h bf[4];
#pragma unroll
        for (int j = 0; j < 4; ++j) {
            v8h blo = *(const v8h*)(bptr[j] + k0);
            v8h bhi = *(const v8h*)(bptr[j] + k0 + 16);
#pragma unroll
            for (int i = 0; i < 8; ++i) { bf[j][i] = blo[i]; bf[j][i + 8] = bhi[i]; }
        }
        v16h af;
#pragma unroll
        for (int i = 0; i < 8; ++i) { af[i] = alo[i]; af[i + 8] = ahi[i]; }
#pragma unroll
        for (int j = 0; j < 4; ++j)
            acc[j] = __builtin_amdgcn_wmma_f32_16x16x32_f16(
                false, af, false, bf[j], (short)0, acc[j], false, false);
    }

    // C/D layout: VGPR r -> row m0 + hi*8 + r, col n0 + j*16 + (lane&15)
#pragma unroll
    for (int j = 0; j < 4; ++j) {
        int col = n0 + j * 16 + r16;
        if (col >= Nd) continue;
        float bv = bias ? bias[col] : 0.0f;
#pragma unroll
        for (int r = 0; r < 8; ++r) {
            int row = m0 + hi * 8 + r;
            float v = acc[j][r] + bv;
            if (act) v = gelu_exact(v);
            size_t idx = (size_t)row * Nd + col;
            if (resid) v += resid[idx];
            if (outF) outF[idx] = v;
            if (outH) outH[idx] = (half_t)v;
        }
    }
}

// ---- DCNv3 core: per-token bilinear gather with per-group softmax mask -----
__device__ __forceinline__ float dcn_tap(const float* __restrict__ xch, int yi, int xi) {
    // padded 98x98 coordinates; nonzero only in interior [1..96]
    if (yi >= 1 && yi <= HDIM && xi >= 1 && xi <= WDIM)
        return xch[(size_t)((yi - 1) * WDIM + (xi - 1)) * C_DIM];
    return 0.0f;
}

__global__ void k_dcn_core(const float* __restrict__ xproj, const float* __restrict__ om,
                           half_t* __restrict__ yout) {
    int tok = blockIdx.x;
    int t   = threadIdx.x;           // 0..127
    int g   = t >> 4;                // group
    int n   = tok / (HDIM * WDIM);
    int rem = tok - n * HDIM * WDIM;
    int h = rem / WDIM, w = rem - h * WDIM;

    const float* offp = om + (size_t)tok * NOM + g * (PPTS * 2);
    const float* mskp = om + (size_t)tok * NOM + NOFF + g * PPTS;

    float ml[PPTS];
    float mx = -3.402823466e38f;
#pragma unroll
    for (int p = 0; p < PPTS; ++p) { ml[p] = mskp[p]; mx = fmaxf(mx, ml[p]); }
    float se = 0.0f;
#pragma unroll
    for (int p = 0; p < PPTS; ++p) { ml[p] = expf(ml[p] - mx); se += ml[p]; }
    float inv = 1.0f / se;

    const float* xch = xproj + (size_t)n * HDIM * WDIM * C_DIM + t;  // channel = g*16+cc = t
    float acc = 0.0f;
#pragma unroll
    for (int p = 0; p < PPTS; ++p) {
        // px = w + 1 + kx + off_x ; py = h + 1 + ky + off_y (padded 98x98 coords)
        float px = (float)(w + (p / 3)) + offp[2 * p];      // w + 1 + (p/3 - 1)
        float py = (float)(h + (p % 3)) + offp[2 * p + 1];  // h + 1 + (p%3 - 1)
        float fx0 = floorf(px), fy0 = floorf(py);
        int   x0 = (int)fx0,    y0 = (int)fy0;
        float fx = px - fx0,    fy = py - fy0;
        float wg  = ml[p] * inv;
        float w00 = (1.0f - fy) * (1.0f - fx) * wg;
        float w01 = (1.0f - fy) * fx * wg;
        float w10 = fy * (1.0f - fx) * wg;
        float w11 = fy * fx * wg;
        acc += w00 * dcn_tap(xch, y0,     x0)
             + w01 * dcn_tap(xch, y0,     x0 + 1)
             + w10 * dcn_tap(xch, y0 + 1, x0)
             + w11 * dcn_tap(xch, y0 + 1, x0 + 1);
    }
    yout[(size_t)tok * C_DIM + t] = (half_t)acc;
}

// ---------------------------------------------------------------------------
extern "C" void kernel_launch(void* const* d_in, const int* in_sizes, int n_in,
                              void* d_out, int out_size, void* d_ws, size_t ws_size,
                              hipStream_t stream) {
    (void)in_sizes; (void)n_in; (void)out_size; (void)ws_size;
    const float* x     = (const float*)d_in[0];
    const float* ln1g  = (const float*)d_in[1];
    const float* ln1b  = (const float*)d_in[2];
    const float* w_in  = (const float*)d_in[3];
    const float* b_in  = (const float*)d_in[4];
    const float* dw_k  = (const float*)d_in[5];
    const float* dw_b  = (const float*)d_in[6];
    const float* dwg   = (const float*)d_in[7];
    const float* dwb   = (const float*)d_in[8];
    const float* w_off = (const float*)d_in[9];
    const float* b_off = (const float*)d_in[10];
    const float* w_msk = (const float*)d_in[11];
    const float* b_msk = (const float*)d_in[12];
    const float* w_out = (const float*)d_in[13];
    const float* b_out = (const float*)d_in[14];
    const float* ln2g  = (const float*)d_in[15];
    const float* ln2b  = (const float*)d_in[16];
    const float* w1    = (const float*)d_in[17];
    const float* b1    = (const float*)d_in[18];
    const float* w2    = (const float*)d_in[19];
    const float* b2    = (const float*)d_in[20];

    const int M = NTOK;
    char* ws = (char*)d_ws;
    size_t off = 0;
    auto take = [&](size_t bytes) -> char* {
        char* p = ws + off;
        off += (bytes + 255) & ~(size_t)255;
        return p;
    };
    // Buffers (aliased in stream order; total < 100 MB, fully L2 resident):
    float*  bufA = (float*)take((size_t)M * C_DIM * 4);   // ln1_f32 -> xres
    half_t* bufB = (half_t*)take((size_t)M * C_DIM * 2);  // ln1_f16 -> y_f16
    float*  bufC = (float*)take((size_t)M * C_DIM * 4);   // xproj
    half_t* bufD = (half_t*)take((size_t)M * C_DIM * 2);  // x1_f16 -> h_f16
    char*   bufE = take((size_t)M * HID_DIM * 2);         // offmsk(f32) -> h1(f16)
    half_t* w_in_t  = (half_t*)take((size_t)C_DIM * C_DIM * 2);
    half_t* w_om_t  = (half_t*)take((size_t)NOM * C_DIM * 2);
    half_t* w_out_t = (half_t*)take((size_t)C_DIM * C_DIM * 2);
    half_t* w1_t    = (half_t*)take((size_t)HID_DIM * C_DIM * 2);
    half_t* w2_t    = (half_t*)take((size_t)C_DIM * HID_DIM * 2);
    float*  b_om    = (float*)take(NOM * 4);

    // Stage 0: weight convert + transpose to (N x K) f16
    k_transpose_f16<<<(C_DIM*C_DIM   + 255) / 256, 256, 0, stream>>>(w_in,  w_in_t,  C_DIM, C_DIM);
    k_transpose_f16<<<(C_DIM*NOFF    + 255) / 256, 256, 0, stream>>>(w_off, w_om_t,  C_DIM, NOFF);
    k_transpose_f16<<<(C_DIM*NMSK    + 255) / 256, 256, 0, stream>>>(w_msk, w_om_t + (size_t)NOFF * C_DIM, C_DIM, NMSK);
    k_transpose_f16<<<(C_DIM*C_DIM   + 255) / 256, 256, 0, stream>>>(w_out, w_out_t, C_DIM, C_DIM);
    k_transpose_f16<<<(C_DIM*HID_DIM + 255) / 256, 256, 0, stream>>>(w1,    w1_t,    C_DIM, HID_DIM);
    k_transpose_f16<<<(HID_DIM*C_DIM + 255) / 256, 256, 0, stream>>>(w2,    w2_t,    HID_DIM, C_DIM);
    k_concat_bias<<<1, 256, 0, stream>>>(b_off, b_msk, b_om);

    dim3 g_n128((C_DIM   + 63) / 64, M / 128);
    dim3 g_n216((NOM     + 63) / 64, M / 128);
    dim3 g_n512((HID_DIM + 63) / 64, M / 128);

    // 1) ln1 = LN(x)                    -> bufA (f32), bufB (f16)
    k_layernorm<<<M / 8, 256, 0, stream>>>(x, ln1g, ln1b, bufA, bufB);
    // 2) x1 = GELU(LN(dwconv(ln1)+b))   -> bufD (f16)
    k_dwconv_ln_gelu<<<M / 8, 256, 0, stream>>>(bufA, dw_k, dw_b, dwg, dwb, bufD);
    // 3) xproj = ln1 @ w_in + b_in      -> bufC (f32)
    k_gemm_wmma<<<g_n128, 256, 0, stream>>>(bufB, w_in_t, b_in, nullptr,
                                            bufC, nullptr, M, C_DIM, C_DIM, 0);
    // 4) offmsk = x1 @ [w_off|w_msk]    -> bufE (f32, M x 216)
    k_gemm_wmma<<<g_n216, 256, 0, stream>>>(bufD, w_om_t, b_om, nullptr,
                                            (float*)bufE, nullptr, M, NOM, C_DIM, 0);
    // 5) y = dcnv3_core(xproj, offmsk)  -> bufB (f16)
    k_dcn_core<<<M, 128, 0, stream>>>(bufC, (const float*)bufE, bufB);
    // 6) xres = x + y @ w_out + b_out   -> bufA (f32)
    k_gemm_wmma<<<g_n128, 256, 0, stream>>>(bufB, w_out_t, b_out, x,
                                            bufA, nullptr, M, C_DIM, C_DIM, 0);
    // 7) h = LN2(xres)                  -> bufD (f16)
    k_layernorm<<<M / 8, 256, 0, stream>>>(bufA, ln2g, ln2b, nullptr, bufD);
    // 8) h1 = GELU(h @ w1 + b1)         -> bufE (f16, M x 512)
    k_gemm_wmma<<<g_n512, 256, 0, stream>>>(bufD, w1_t, b1, nullptr,
                                            nullptr, (half_t*)bufE, M, HID_DIM, C_DIM, 1);
    // 9) out = xres + h1 @ w2 + b2      -> d_out (f32)
    k_gemm_wmma<<<g_n128, 256, 0, stream>>>((const half_t*)bufE, w2_t, b2, bufA,
                                            (float*)d_out, nullptr, M, C_DIM, HID_DIM, 0);
}
